// GroupedExperts_82678120448634
// MI455X (gfx1250) — matrix operand order repaired and checked by reading, not compiled, not run on hardware
//
#include <hip/hip_runtime.h>

// ---------------------------------------------------------------------------
// MoE grouped-GEMM (SwiGLU FFN) for gfx1250 / MI455X.
//   GEMM1: x1 = X @ W1^T, x3 = X @ W3^T  (fused, shared A fragments)
//   h    = silu(x1) * x3 * routed_prob   (stored bf16 in workspace)
//   GEMM2: out = h @ W2^T                (f32 output)
// bf16 WMMA 16x16x32 w/ f32 accum; double-buffered LDS, one barrier per
// K-step; GEMM2 A-tile staged with global_load_async_to_lds_b128 (ASYNCcnt).
// ---------------------------------------------------------------------------

#define E_ 8
#define D_ 2048
#define H_ 1408
#define T_ 16384

#define TM 128          // token tile (M)
#define TN 64           // output-col tile (N)
#define TK 32           // K step (WMMA K)
#define LDSK (TK + 8)   // padded LDS row stride in bf16 elems (80B, 16B aligned)

typedef __attribute__((ext_vector_type(16))) __bf16        v16bf;
typedef __attribute__((ext_vector_type(2)))  __bf16        v2bf;
typedef __attribute__((ext_vector_type(8)))  float         v8f;
typedef __attribute__((ext_vector_type(4)))  unsigned int  v4u;

union FragU { v4u q[2]; v16bf v; };

// f32 -> bf16 pair; lowers to v_cvt_pk_bf16_f32 (verified in round-3 asm).
static __device__ __forceinline__ unsigned int pack_bf16x2(float lo, float hi) {
    v2bf p;
    p.x = (__bf16)lo;
    p.y = (__bf16)hi;
    unsigned int r;
    __builtin_memcpy(&r, &p, 4);
    return r;
}

static __device__ __forceinline__ unsigned short bf16_bits(float f) {
    __bf16 b = (__bf16)f;
    unsigned short r;
    __builtin_memcpy(&r, &b, 2);
    return r;
}

static __device__ __forceinline__ float fast_rcp(float x) {
#if __has_builtin(__builtin_amdgcn_rcpf)
    return __builtin_amdgcn_rcpf(x);
#else
    return 1.0f / x;
#endif
}

// Pack 8 consecutive f32 (two float4) into 4 bf16x2 dwords for ds_store_b128.
static __device__ __forceinline__ v4u pack8(const float4& a, const float4& b) {
    v4u q;
    q.x = pack_bf16x2(a.x, a.y);
    q.y = pack_bf16x2(a.z, a.w);
    q.z = pack_bf16x2(b.x, b.y);
    q.w = pack_bf16x2(b.z, b.w);
    return q;
}

// Async DMA: 16B global -> LDS, tracked on ASYNCcnt. LDS byte address is the
// low 32 bits of the generic pointer.
static __device__ __forceinline__ void async_copy_b128(const void* gsrc,
                                                       const unsigned short* lds_dst) {
    unsigned int loff = (unsigned int)(unsigned long long)(uintptr_t)lds_dst;
    asm volatile("global_load_async_to_lds_b128 %0, %1, off"
                 :: "v"(loff), "v"(gsrc)
                 : "memory");
}
static __device__ __forceinline__ void wait_async0() {
    asm volatile("s_wait_asynccnt 0x0" ::: "memory");
}

// A fragment: 16x32 bf16, rows M. Lane 0-15 -> M=lane, K in {0..7,16..23};
// lane 16-31 -> M=lane-16, K in {8..15,24..31}. Two ds_read_b128 per frag.
static __device__ __forceinline__ v16bf load_frag_A(const unsigned short* lds,
                                                    int row_base, int lane) {
    const int row = row_base + (lane & 15);
    const int kb  = ((lane >> 4) * 4) * 4;
    const char* p = (const char*)(lds + row * LDSK);
    FragU f;
    f.q[0] = *(const v4u*)(p + kb);
    f.q[1] = *(const v4u*)(p + kb + 32);
    return f.v;
}

// B fragment: 32x16 bf16 (K x N), LDS layout [N][K] row-major.
// Lane 0-15 -> N=lane, K in {0..7,16..23}; lane 16-31 -> N=lane-16, K in {8..15,24..31}.
static __device__ __forceinline__ v16bf load_frag_B(const unsigned short* lds,
                                                    int col_base, int lane) {
    const int row = col_base + (lane & 15);
    const int kb  = ((lane >> 4) * 4) * 4;
    const char* p = (const char*)(lds + row * LDSK);
    FragU f;
    f.q[0] = *(const v4u*)(p + kb);
    f.q[1] = *(const v4u*)(p + kb + 32);
    return f.v;
}

static __device__ __forceinline__ int expert_of_tile(const int* __restrict__ ntok,
                                                     int m0) {
    int e = 0, off = 0;
    #pragma unroll
    for (int i = 0; i < E_; ++i) {
        int c = ntok[i];
        if (m0 >= off + c) { off += c; e = i + 1; }
    }
    return e;
}

// ---------------------------------------------------------------------------
// GEMM1 + fused SwiGLU * routed_prob.  grid = (H/64, T/128), block = 256.
// ---------------------------------------------------------------------------
__global__ __launch_bounds__(256)
void moe_gemm1_swiglu(const float* __restrict__ x,
                      const float* __restrict__ w1,
                      const float* __restrict__ w3,
                      const float* __restrict__ prob,
                      const int*   __restrict__ ntok,
                      unsigned short* __restrict__ hbuf /* bf16 [T,H] */) {
    __shared__ unsigned short lA [2 * TM * LDSK];
    __shared__ unsigned short lB1[2 * TN * LDSK];
    __shared__ unsigned short lB3[2 * TN * LDSK];

    const int m0 = blockIdx.y * TM;
    const int n0 = blockIdx.x * TN;
    const int e  = expert_of_tile(ntok, m0);

    const float* W1 = w1 + (size_t)e * H_ * D_;   // [H][D] rows = N, cols = K
    const float* W3 = w3 + (size_t)e * H_ * D_;

    const int tid  = threadIdx.x;
    const int lane = tid & 31;
    const int wave = tid >> 5;
    const int wm   = wave & 3;    // 4 M-subtiles of 32 rows
    const int wn   = wave >> 2;   // 2 N-subtiles of 32 cols

    // Per-thread 8-float chunk ownership (row, col8) per tile:
    //   A (128x32): 512 chunks -> 2 per thread;  B (64x32): 256 -> 1 per thread.
    const int arow0 = tid >> 2,          ac8 = (tid & 3) * 8;
    const int arow1 = (tid + 256) >> 2;  // second A chunk, same ac8
    const int brow  = tid >> 2,          bc8 = (tid & 3) * 8;

    v8f acc1[2][2], acc3[2][2];
    #pragma unroll
    for (int i = 0; i < 2; ++i)
        #pragma unroll
        for (int j = 0; j < 2; ++j) { acc1[i][j] = {}; acc3[i][j] = {}; }

    float4 rA[4], rB1[2], rB3[2];   // staged global data (8 floats per chunk)

    auto glob_load = [&](int k0) {
        const float* pa0 = x + (size_t)(m0 + arow0) * D_ + k0 + ac8;
        const float* pa1 = x + (size_t)(m0 + arow1) * D_ + k0 + ac8;
        rA[0] = *(const float4*)(pa0);
        rA[1] = *(const float4*)(pa0 + 4);
        rA[2] = *(const float4*)(pa1);
        rA[3] = *(const float4*)(pa1 + 4);
        const size_t go = (size_t)(n0 + brow) * D_ + k0 + bc8;
        rB1[0] = *(const float4*)(W1 + go);
        rB1[1] = *(const float4*)(W1 + go + 4);
        rB3[0] = *(const float4*)(W3 + go);
        rB3[1] = *(const float4*)(W3 + go + 4);
    };

    glob_load(0);
    int buf = 0;
    for (int k0 = 0; k0 < D_; k0 += TK) {
        unsigned short* sA  = lA  + buf * (TM * LDSK);
        unsigned short* sB1 = lB1 + buf * (TN * LDSK);
        unsigned short* sB3 = lB3 + buf * (TN * LDSK);

        *(v4u*)&sA [arow0 * LDSK + ac8] = pack8(rA[0], rA[1]);
        *(v4u*)&sA [arow1 * LDSK + ac8] = pack8(rA[2], rA[3]);
        *(v4u*)&sB1[brow  * LDSK + bc8] = pack8(rB1[0], rB1[1]);
        *(v4u*)&sB3[brow  * LDSK + bc8] = pack8(rB3[0], rB3[1]);
        __syncthreads();                 // stage visible to all waves
        if (k0 + TK < D_) glob_load(k0 + TK);   // next tile: loads in flight

        const v16bf a0  = load_frag_A(sA,  wm * 32 + 0,  lane);
        const v16bf a1  = load_frag_A(sA,  wm * 32 + 16, lane);
        const v16bf b10 = load_frag_B(sB1, wn * 32 + 0,  lane);
        const v16bf b11 = load_frag_B(sB1, wn * 32 + 16, lane);
        const v16bf b30 = load_frag_B(sB3, wn * 32 + 0,  lane);
        const v16bf b31 = load_frag_B(sB3, wn * 32 + 16, lane);

        acc1[0][0] = __builtin_amdgcn_wmma_f32_16x16x32_bf16(false, a0, false, b10, (short)0, acc1[0][0], false, false);
        acc1[0][1] = __builtin_amdgcn_wmma_f32_16x16x32_bf16(false, a0, false, b11, (short)0, acc1[0][1], false, false);
        acc1[1][0] = __builtin_amdgcn_wmma_f32_16x16x32_bf16(false, a1, false, b10, (short)0, acc1[1][0], false, false);
        acc1[1][1] = __builtin_amdgcn_wmma_f32_16x16x32_bf16(false, a1, false, b11, (short)0, acc1[1][1], false, false);
        acc3[0][0] = __builtin_amdgcn_wmma_f32_16x16x32_bf16(false, a0, false, b30, (short)0, acc3[0][0], false, false);
        acc3[0][1] = __builtin_amdgcn_wmma_f32_16x16x32_bf16(false, a0, false, b31, (short)0, acc3[0][1], false, false);
        acc3[1][0] = __builtin_amdgcn_wmma_f32_16x16x32_bf16(false, a1, false, b30, (short)0, acc3[1][0], false, false);
        acc3[1][1] = __builtin_amdgcn_wmma_f32_16x16x32_bf16(false, a1, false, b31, (short)0, acc3[1][1], false, false);
        buf ^= 1;
    }

    // Epilogue: h = silu(x1) * x3 * prob, store bf16.
    // C/D layout: lane 0-15 -> N=lane, M=r; lane 16-31 -> N=lane-16, M=8+r.
    const int m_base = m0 + wm * 32;
    const int n_base = n0 + wn * 32;
    const int mh     = (lane >> 4) << 3;
    const int nc     = lane & 15;
    #pragma unroll
    for (int mi = 0; mi < 2; ++mi) {
        #pragma unroll
        for (int r = 0; r < 8; ++r) {
            const int token = m_base + mi * 16 + mh + r;
            const float pr  = prob[token];
            #pragma unroll
            for (int ni = 0; ni < 2; ++ni) {
                const float v1 = acc1[mi][ni][r];
                const float v3 = acc3[mi][ni][r];
                const float s  = v1 * fast_rcp(1.0f + __expf(-v1));
                const float hv = s * v3 * pr;
                const int col  = n_base + ni * 16 + nc;
                hbuf[(size_t)token * H_ + col] = bf16_bits(hv);
            }
        }
    }
}

// ---------------------------------------------------------------------------
// GEMM2: out = h @ W2^T.  grid = (D/64, T/128), block = 256.
// A tile staged by async LDS-DMA (data already bf16); B via cvt pipeline.
// ---------------------------------------------------------------------------
__global__ __launch_bounds__(256)
void moe_gemm2(const unsigned short* __restrict__ hbuf /* bf16 [T,H] */,
               const float* __restrict__ w2,
               const int*   __restrict__ ntok,
               float* __restrict__ out /* f32 [T,D] */) {
    __shared__ unsigned short lA[2 * TM * LDSK];
    __shared__ unsigned short lB[2 * TN * LDSK];

    const int m0 = blockIdx.y * TM;
    const int n0 = blockIdx.x * TN;
    const int e  = expert_of_tile(ntok, m0);

    const float* W2 = w2 + (size_t)e * D_ * H_;   // [D][H] rows = N, cols = K

    const int tid  = threadIdx.x;
    const int lane = tid & 31;
    const int wave = tid >> 5;
    const int wm   = wave & 3;
    const int wn   = wave >> 2;

    const int arow0 = tid >> 2,          ac8 = (tid & 3) * 8;
    const int arow1 = (tid + 256) >> 2;
    const int brow  = tid >> 2,          bc8 = (tid & 3) * 8;

    v8f acc[2][2];
    #pragma unroll
    for (int i = 0; i < 2; ++i)
        #pragma unroll
        for (int j = 0; j < 2; ++j) acc[i][j] = {};

    float4 rB[2];    // f32 weights: one 8-float chunk -> convert

    auto async_A = [&](int k0, unsigned short* sA) {
        async_copy_b128(hbuf + (size_t)(m0 + arow0) * H_ + k0 + ac8,
                        &sA[arow0 * LDSK + ac8]);
        async_copy_b128(hbuf + (size_t)(m0 + arow1) * H_ + k0 + ac8,
                        &sA[arow1 * LDSK + ac8]);
    };
    auto glob_B = [&](int k0) {
        const float* pb = W2 + (size_t)(n0 + brow) * H_ + k0 + bc8;
        rB[0] = *(const float4*)(pb);
        rB[1] = *(const float4*)(pb + 4);
    };

    async_A(0, lA);
    glob_B(0);
    int buf = 0;
    for (int k0 = 0; k0 < H_; k0 += TK) {
        unsigned short* sA = lA + buf * (TM * LDSK);
        unsigned short* sB = lB + buf * (TN * LDSK);

        *(v4u*)&sB[brow * LDSK + bc8] = pack8(rB[0], rB[1]);
        wait_async0();                   // this wave's A-DMA into sA complete
        __syncthreads();                 // everyone's fill + DMA visible
        if (k0 + TK < H_) {              // next tile in flight over the WMMAs
            async_A(k0 + TK, lA + (buf ^ 1) * (TM * LDSK));
            glob_B(k0 + TK);
        }

        const v16bf a0 = load_frag_A(sA, wm * 32 + 0,  lane);
        const v16bf a1 = load_frag_A(sA, wm * 32 + 16, lane);
        const v16bf b0 = load_frag_B(sB, wn * 32 + 0,  lane);
        const v16bf b1 = load_frag_B(sB, wn * 32 + 16, lane);

        acc[0][0] = __builtin_amdgcn_wmma_f32_16x16x32_bf16(false, a0, false, b0, (short)0, acc[0][0], false, false);
        acc[0][1] = __builtin_amdgcn_wmma_f32_16x16x32_bf16(false, a0, false, b1, (short)0, acc[0][1], false, false);
        acc[1][0] = __builtin_amdgcn_wmma_f32_16x16x32_bf16(false, a1, false, b0, (short)0, acc[1][0], false, false);
        acc[1][1] = __builtin_amdgcn_wmma_f32_16x16x32_bf16(false, a1, false, b1, (short)0, acc[1][1], false, false);
        buf ^= 1;
    }

    const int m_base = m0 + wm * 32;
    const int n_base = n0 + wn * 32;
    const int mh     = (lane >> 4) << 3;
    const int nc     = lane & 15;
    #pragma unroll
    for (int mi = 0; mi < 2; ++mi)
        #pragma unroll
        for (int ni = 0; ni < 2; ++ni)
            #pragma unroll
            for (int r = 0; r < 8; ++r) {
                const int token = m_base + mi * 16 + mh + r;
                const int col   = n_base + ni * 16 + nc;
                out[(size_t)token * D_ + col] = acc[mi][ni][r];
            }
}

// ---------------------------------------------------------------------------
extern "C" void kernel_launch(void* const* d_in, const int* in_sizes, int n_in,
                              void* d_out, int out_size, void* d_ws, size_t ws_size,
                              hipStream_t stream) {
    const float* x    = (const float*)d_in[0];
    const float* w1   = (const float*)d_in[1];
    const float* w2   = (const float*)d_in[2];
    const float* w3   = (const float*)d_in[3];
    const float* prob = (const float*)d_in[4];
    const int*   ntok = (const int*)d_in[5];

    unsigned short* hbuf = (unsigned short*)d_ws;   // bf16 [T,H] = 46 MB
    float* out = (float*)d_out;

    dim3 blk(256);
    moe_gemm1_swiglu<<<dim3(H_ / TN, T_ / TM), blk, 0, stream>>>(x, w1, w3, prob, ntok, hbuf);
    moe_gemm2      <<<dim3(D_ / TN, T_ / TM), blk, 0, stream>>>(hbuf, w2, ntok, out);
}
